// FactoredLSTM_83425444757637
// MI455X (gfx1250) — matrix operand order repaired and checked by reading, not compiled, not run on hardware
//
#include <hip/hip_runtime.h>
#include <hip/hip_bf16.h>
#include <math.h>

// ---------------------------------------------------------------------------
// Problem dimensions (from the reference)
// ---------------------------------------------------------------------------
static constexpr int kB  = 64;      // batch
static constexpr int kT  = 30;      // timesteps
static constexpr int kE  = 300;     // embedding dim
static constexpr int kEp = 320;     // embedding dim padded to multiple of 32
static constexpr int kH  = 512;     // hidden (= FD)
static constexpr int kV  = 10000;   // vocab
static constexpr int kBT = kB * kT; // 1920 rows for the batched GEMMs

// ---------------------------------------------------------------------------
// WMMA vector types (gfx1250, wave32)
// ---------------------------------------------------------------------------
typedef __attribute__((ext_vector_type(16))) __bf16 v16bf;
typedef __attribute__((ext_vector_type(8)))  __bf16 v8bf;
typedef __attribute__((ext_vector_type(8)))  float  v8f;

// ---------------------------------------------------------------------------
// One wave computes an MB x NB block of 16x16 f32 tiles of
//   C = A(bf16[M,K]) * B^T(bf16[N,K]) + bias
// A row-major [M,K], K % 32 == 0.  Bt row-major [N,K] (i.e. B transposed).
// Fragment layouts per CDNA5 ISA 7.12.2:
//   A frag : lane L -> row m = L%16 ; elem i -> k = (i/8)*16 + (L/16)*8 + i%8
//   B frag : lane L -> col n = L%16 ; elem i -> k = (i/8)*16 + (L/16)*8 + i%8
//   C frag : lane L -> col n = L%16 ; vgpr r -> row m = (L/16)*8 + r
// Register blocking reuses each A fragment NB times and each B fragment MB
// times: (2*MB + 2*NB) b128 loads feed MB*NB WMMAs per k-step.
// ---------------------------------------------------------------------------
template <int MB, int NB, bool OUT_BF16>
__device__ __forceinline__ void wave_gemm_block(
    const __bf16* __restrict__ A, const __bf16* __restrict__ Bt,
    const float* __restrict__ bias, void* __restrict__ out,
    int N, int K, int tile_m0, int tile_n0, int lane)
{
    const int col  = lane & 15;
    const int half = lane >> 4;

    const __bf16* arow[MB];
    const __bf16* brow[NB];
#pragma unroll
    for (int mi = 0; mi < MB; ++mi)
        arow[mi] = A + (size_t)((tile_m0 + mi) * 16 + col) * K + half * 8;
#pragma unroll
    for (int ni = 0; ni < NB; ++ni)
        brow[ni] = Bt + (size_t)((tile_n0 + ni) * 16 + col) * K + half * 8;

    v8f acc[MB][NB];
#pragma unroll
    for (int mi = 0; mi < MB; ++mi)
#pragma unroll
        for (int ni = 0; ni < NB; ++ni)
            acc[mi][ni] = (v8f){};

    for (int k0 = 0; k0 < K; k0 += 32) {
        v16bf a[MB], b[NB];
#pragma unroll
        for (int mi = 0; mi < MB; ++mi) {
            v8bf alo = *(const v8bf*)(arow[mi] + k0);       // k = k0 + half*8 + [0..7]
            v8bf ahi = *(const v8bf*)(arow[mi] + k0 + 16);  // k = k0 + 16 + half*8 + [0..7]
#pragma unroll
            for (int i = 0; i < 8; ++i) { a[mi][i] = alo[i]; a[mi][i + 8] = ahi[i]; }
        }
#pragma unroll
        for (int ni = 0; ni < NB; ++ni) {
            v8bf blo = *(const v8bf*)(brow[ni] + k0);       // k = k0 + half*8 + [0..7]
            v8bf bhi = *(const v8bf*)(brow[ni] + k0 + 16);  // k = k0 + 16 + half*8 + [0..7]
#pragma unroll
            for (int i = 0; i < 8; ++i) { b[ni][i] = blo[i]; b[ni][i + 8] = bhi[i]; }
        }

#pragma unroll
        for (int mi = 0; mi < MB; ++mi)
#pragma unroll
            for (int ni = 0; ni < NB; ++ni)
                acc[mi][ni] = __builtin_amdgcn_wmma_f32_16x16x32_bf16(
                    /*neg_a=*/false, a[mi], /*neg_b=*/false, b[ni],
                    /*c_mod=*/(short)0, acc[mi][ni],
                    /*reuse_a=*/false, /*reuse_b=*/false);
    }

#pragma unroll
    for (int ni = 0; ni < NB; ++ni) {
        const int   n  = (tile_n0 + ni) * 16 + col;
        const float bv = bias ? bias[n] : 0.0f;
#pragma unroll
        for (int mi = 0; mi < MB; ++mi) {
#pragma unroll
            for (int r = 0; r < 8; ++r) {
                const int   m = (tile_m0 + mi) * 16 + half * 8 + r;
                const float v = acc[mi][ni][r] + bv;
                if (OUT_BF16) ((__bf16*)out)[(size_t)m * N + n] = (__bf16)v;
                else          ((float*)out)[(size_t)m * N + n]  = v;
            }
        }
    }
}

template <int MB, int NB, bool OUT_BF16>
__global__ void __launch_bounds__(32) gemm16(
    const __bf16* __restrict__ A, const __bf16* __restrict__ Bt,
    const float* __restrict__ bias, void* __restrict__ out, int N, int K)
{
    wave_gemm_block<MB, NB, OUT_BF16>(A, Bt, bias, out, N, K,
                                      (int)blockIdx.y * MB, (int)blockIdx.x * NB,
                                      (int)threadIdx.x);
}

// Fused 4-gate recurrent GEMM:  G[g] = h @ W_g + W_g_b   (M=64, N=K=512)
__global__ void __launch_bounds__(32) gate_gemm(
    const __bf16* __restrict__ hbf, const __bf16* __restrict__ WtAll,
    const float* __restrict__ bi, const float* __restrict__ bf_,
    const float* __restrict__ bo, const float* __restrict__ bc,
    float* __restrict__ Gall)
{
    const int g = blockIdx.z;
    const float* bias = (g == 0) ? bi : (g == 1) ? bf_ : (g == 2) ? bo : bc;
    wave_gemm_block<2, 4, false>(hbf, WtAll + (size_t)g * kH * kH, bias,
                                 Gall + (size_t)g * kB * kH, kH, kH,
                                 (int)blockIdx.y * 2, (int)blockIdx.x * 4,
                                 (int)threadIdx.x);
}

// ---------------------------------------------------------------------------
// Data preparation kernels
// ---------------------------------------------------------------------------

// LDS-tiled transpose-cast: in fp32 [K,N] -> out bf16 [N,Kp] (K zero-padded).
// Both the fp32 reads and the bf16 writes are coalesced; 16x17 LDS tile avoids
// bank conflicts. Requires N % 16 == 0 and Kp % 16 == 0 (true for all uses).
__global__ void __launch_bounds__(256) transpose_cast(
    const float* __restrict__ in, __bf16* __restrict__ out, int K, int N, int Kp)
{
    __shared__ float tile[16][17];
    const int tx = threadIdx.x, ty = threadIdx.y;
    const int n0 = blockIdx.x * 16;
    const int k0 = blockIdx.y * 16;

    const int k = k0 + ty;
    tile[ty][tx] = (k < K) ? in[(size_t)k * N + (n0 + tx)] : 0.0f;
    __syncthreads();
    out[(size_t)(n0 + ty) * Kp + (k0 + tx)] = (__bf16)tile[tx][ty];
}

// X[b,t,:] = (t==0) ? features[b] : B_emb[captions[b,t-1]] ; bf16, E padded to 320
__global__ void build_x(const int* __restrict__ captions,
                        const float* __restrict__ features,
                        const float* __restrict__ Bemb,
                        __bf16* __restrict__ Xbf)
{
    int idx = blockIdx.x * blockDim.x + threadIdx.x;
    if (idx >= kBT * kEp) return;
    const int e  = idx % kEp;
    const int bt = idx / kEp;
    const int t  = bt % kT;
    const int b  = bt / kT;
    float v = 0.0f;
    if (e < kE) {
        v = (t == 0) ? features[(size_t)b * kE + e]
                     : Bemb[(size_t)captions[b * kT + (t - 1)] * kE + e];
    }
    Xbf[idx] = (__bf16)v;
}

__global__ void init_hc(const float* __restrict__ h0, const float* __restrict__ c0,
                        __bf16* __restrict__ hbf, float* __restrict__ c)
{
    int idx = blockIdx.x * blockDim.x + threadIdx.x;
    if (idx >= kB * kH) return;
    hbf[idx] = (__bf16)h0[idx];
    c[idx]   = c0[idx];
}

// Elementwise LSTM cell update for timestep t. Writes c, next-h (bf16) and
// Hs[b*T+t] (bf16 operand rows for the final vocab projection).
__global__ void lstm_step(const float* __restrict__ Uall,  // [4][B*T, H]
                          const float* __restrict__ Gall,  // [4][B, H]
                          float* __restrict__ c,
                          __bf16* __restrict__ hbf,
                          __bf16* __restrict__ Hs,          // [B*T, H]
                          int t)
{
    int idx = blockIdx.x * blockDim.x + threadIdx.x;
    if (idx >= kB * kH) return;
    const int b  = idx / kH;
    const int hh = idx % kH;
    const size_t mrow = (size_t)(b * kT + t) * kH + hh;
    const size_t gU   = (size_t)kBT * kH;
    const size_t gG   = (size_t)kB * kH;

    const float gi = Uall[0 * gU + mrow] + Gall[0 * gG + idx];
    const float gf = Uall[1 * gU + mrow] + Gall[1 * gG + idx];
    const float go = Uall[2 * gU + mrow] + Gall[2 * gG + idx];
    const float gc = Uall[3 * gU + mrow] + Gall[3 * gG + idx];

    const float i_t = 1.0f / (1.0f + expf(-gi));
    const float f_t = 1.0f / (1.0f + expf(-gf));
    const float o_t = 1.0f / (1.0f + expf(-go));
    const float ct  = tanhf(gc);

    const float cn = f_t * c[idx] + i_t * ct;
    const float hn = o_t * cn;                 // note: no tanh, as in reference

    c[idx]   = cn;
    hbf[idx] = (__bf16)hn;
    Hs[mrow] = (__bf16)hn;
}

// ---------------------------------------------------------------------------
// Host orchestration
// ---------------------------------------------------------------------------
extern "C" void kernel_launch(void* const* d_in, const int* in_sizes, int n_in,
                              void* d_out, int out_size, void* d_ws, size_t ws_size,
                              hipStream_t stream)
{
    (void)in_sizes; (void)n_in; (void)out_size; (void)ws_size;

    const int*   captions = (const int*)  d_in[0];
    const float* features = (const float*)d_in[1];
    const float* h0       = (const float*)d_in[2];
    const float* c0       = (const float*)d_in[3];
    const float* Bemb     = (const float*)d_in[4];
    const float* C_W      = (const float*)d_in[37];
    const float* C_b      = (const float*)d_in[38];

    // Workspace carve-out (256B aligned)
    char*  ws  = (char*)d_ws;
    size_t off = 0;
    auto alloc = [&](size_t bytes) -> void* {
        void* p = ws + off;
        off += (bytes + 255) & ~(size_t)255;
        return p;
    };

    __bf16* Xbf   = (__bf16*)alloc((size_t)kBT * kEp * 2);        // [1920,320]
    __bf16* VtAll = (__bf16*)alloc((size_t)4 * kH * kEp * 2);     // 4 x [512,320]
    __bf16* StAll = (__bf16*)alloc((size_t)4 * kH * kH * 2);      // 4 x [512,512]
    __bf16* UtAll = (__bf16*)alloc((size_t)4 * kH * kH * 2);
    __bf16* WtAll = (__bf16*)alloc((size_t)4 * kH * kH * 2);
    __bf16* Ct    = (__bf16*)alloc((size_t)kV * kH * 2);          // [10000,512]
    __bf16* F1    = (__bf16*)alloc((size_t)kBT * kH * 2);
    __bf16* F2    = (__bf16*)alloc((size_t)kBT * kH * 2);
    float*  Uall  = (float*) alloc((size_t)4 * kBT * kH * 4);     // 4 x [1920,512]
    float*  Gall  = (float*) alloc((size_t)4 * kB * kH * 4);      // 4 x [64,512]
    __bf16* hbf   = (__bf16*)alloc((size_t)kB * kH * 2);
    float*  cbuf  = (float*) alloc((size_t)kB * kH * 4);
    __bf16* Hs    = (__bf16*)alloc((size_t)kBT * kH * 2);         // [1920,512]

    auto cdiv = [](int a, int b) { return (a + b - 1) / b; };
    const dim3 tb(16, 16);

    // ---- 1) Weight transpose-casts to bf16 [N, Kpad] --------------------
    for (int g = 0; g < 4; ++g) {
        const int base = 5 + 8 * g;  // V_W,V_b,S_W,S_b,U_W,U_b,W_W,W_b per gate
        const float* VW = (const float*)d_in[base + 0];
        const float* SW = (const float*)d_in[base + 2];
        const float* UW = (const float*)d_in[base + 4];
        const float* WW = (const float*)d_in[base + 6];
        transpose_cast<<<dim3(kH / 16, kEp / 16), tb, 0, stream>>>(VW, VtAll + (size_t)g * kH * kEp, kE, kH, kEp);
        transpose_cast<<<dim3(kH / 16, kH / 16), tb, 0, stream>>>(SW, StAll + (size_t)g * kH * kH, kH, kH, kH);
        transpose_cast<<<dim3(kH / 16, kH / 16), tb, 0, stream>>>(UW, UtAll + (size_t)g * kH * kH, kH, kH, kH);
        transpose_cast<<<dim3(kH / 16, kH / 16), tb, 0, stream>>>(WW, WtAll + (size_t)g * kH * kH, kH, kH, kH);
    }
    transpose_cast<<<dim3(kV / 16, kH / 16), tb, 0, stream>>>(C_W, Ct, kH, kV, kH);

    // ---- 2) Build X (embedding gather, features at t=0), init h/c ------
    build_x<<<cdiv(kBT * kEp, 256), 256, 0, stream>>>(captions, features, Bemb, Xbf);
    init_hc<<<cdiv(kB * kH, 256), 256, 0, stream>>>(h0, c0, hbf, cbuf);

    // ---- 3) Time-parallel gate pre-activations: U_g = ((X@V)@S)@U ------
    // M=1920 (MB=2 -> 60), N=512 (NB=4 -> 8)
    for (int g = 0; g < 4; ++g) {
        const int base = 5 + 8 * g;
        const float* Vb = (const float*)d_in[base + 1];
        const float* Sb = (const float*)d_in[base + 3];
        const float* Ub = (const float*)d_in[base + 5];
        gemm16<2, 4, true><<<dim3(kH / 64, kBT / 32), 32, 0, stream>>>(
            Xbf, VtAll + (size_t)g * kH * kEp, Vb, F1, kH, kEp);
        gemm16<2, 4, true><<<dim3(kH / 64, kBT / 32), 32, 0, stream>>>(
            F1, StAll + (size_t)g * kH * kH, Sb, F2, kH, kH);
        gemm16<2, 4, false><<<dim3(kH / 64, kBT / 32), 32, 0, stream>>>(
            F2, UtAll + (size_t)g * kH * kH, Ub, Uall + (size_t)g * kBT * kH, kH, kH);
    }

    // ---- 4) Sequential recurrence ---------------------------------------
    const float* Wbi = (const float*)d_in[5 + 8 * 0 + 7];
    const float* Wbf = (const float*)d_in[5 + 8 * 1 + 7];
    const float* Wbo = (const float*)d_in[5 + 8 * 2 + 7];
    const float* Wbc = (const float*)d_in[5 + 8 * 3 + 7];
    for (int t = 0; t < kT; ++t) {
        gate_gemm<<<dim3(kH / 64, kB / 32, 4), 32, 0, stream>>>(
            hbf, WtAll, Wbi, Wbf, Wbo, Wbc, Gall);
        lstm_step<<<cdiv(kB * kH, 256), 256, 0, stream>>>(Uall, Gall, cbuf, hbf, Hs, t);
    }

    // ---- 5) Output projection: out[b,t,:] = Hs[b*T+t] @ C_W + C_b ------
    // M=1920 (MB=2 -> 60), N=10000 (NB=5: 80 cols/wave -> 125 blocks)
    gemm16<2, 5, false><<<dim3(kV / 80, kBT / 32), 32, 0, stream>>>(
        Hs, Ct, C_b, d_out, kV, kH);
}
